// Kanv2d_25812753449690
// MI455X (gfx1250) — compile-verified, matrix operand order checked by reading, not run on hardware
//
#include <hip/hip_runtime.h>
#include <hip/hip_bf16.h>

// ---------------------------------------------------------------------------
// KAN spline conv as implicit GEMM on CDNA5 WMMA (v_wmma_f32_16x16x32_bf16)
//   M = B*H*W = 16384, N = Cout = 64, K = Cin*GS*3*3 = 4608
// ---------------------------------------------------------------------------

typedef __attribute__((ext_vector_type(16))) __bf16 v16bf;
typedef __attribute__((ext_vector_type(8)))  __bf16 v8bf;
typedef __attribute__((ext_vector_type(8)))  float  v8f;
typedef __attribute__((ext_vector_type(4)))  float  v4f;

#define BATCH 16
#define CIN   64
#define COUT  64
#define HH    32
#define WW    32
#define HP    34   // padded
#define WP    34
#define GS    8    // GRID_SIZE + SPLINE_ORDER
#define KPIX  512  // CIN*GS, contiguous K per conv tap

// ---------------------------------------------------------------------------
// Kernel 1: cubic B-spline basis expansion of the zero-padded input.
// One thread per (b, hp, wp, i); writes 8 bf16 bases contiguously so that the
// GEMM K-dimension (i*8+g) is contiguous per padded pixel.
// Layout: bases[ ((b*34+hp)*34+wp)*512 + i*8 + g ]
// ---------------------------------------------------------------------------
__global__ __launch_bounds__(256)
void kan_bases(const float* __restrict__ x, __bf16* __restrict__ bases, int total) {
    int tid = blockIdx.x * blockDim.x + threadIdx.x;
    if (tid >= total) return;
    int i  = tid & 63;
    int p  = tid >> 6;          // (b*34+hp)*34+wp
    int wp = p % 34;
    int t  = p / 34;
    int hp = t % 34;
    int b  = t / 34;

    float xv = 0.0f;            // zero padding
    if (hp >= 1 && hp <= HH && wp >= 1 && wp <= WW)
        xv = x[((b * CIN + i) * HH + (hp - 1)) * WW + (wp - 1)];

    const float h = 0.4f;       // (1 - (-1)) / GRID_SIZE
    float knots[12];
#pragma unroll
    for (int j = 0; j < 12; ++j) knots[j] = -1.0f + (float)(j - 3) * h;

    float bas[11];
#pragma unroll
    for (int j = 0; j < 11; ++j)
        bas[j] = (xv >= knots[j] && xv < knots[j + 1]) ? 1.0f : 0.0f;

#pragma unroll
    for (int k = 1; k <= 3; ++k) {
        float inv = 1.0f / ((float)k * h);   // uniform knots: both denoms = k*h
#pragma unroll
        for (int j = 0; j + k < 11; ++j)     // ascending in-place is safe
            bas[j] = ((xv - knots[j]) * bas[j] +
                      (knots[j + k + 1] - xv) * bas[j + 1]) * inv;
    }

    v8bf outv;
#pragma unroll
    for (int g = 0; g < GS; ++g) outv[g] = (__bf16)bas[g];
    *(v8bf*)(bases + (size_t)tid * GS) = outv;   // one 16B store
}

// ---------------------------------------------------------------------------
// Kernel 2: fuse spline_scaler into spline_weight, cast to bf16, transpose to
// wpk[ tap ][ o ][ i*8+g ]  (K contiguous => B-matrix column-contiguous)
// ---------------------------------------------------------------------------
__global__ __launch_bounds__(256)
void kan_pack_w(const float* __restrict__ sw, const float* __restrict__ sc,
                __bf16* __restrict__ wpk, int total) {
    int tid = blockIdx.x * blockDim.x + threadIdx.x;
    if (tid >= total) return;
    int k   = tid & (KPIX - 1);       // i*8+g
    int o   = (tid >> 9) & 63;
    int tap = tid >> 15;              // kh*3+kw, 0..8
    int i   = k >> 3;
    int g   = k & 7;
    int kh  = tap / 3;
    int kw  = tap % 3;
    // spline_weight: [O, I, GS, 3, 3]
    float v = sw[(((o * CIN + i) * GS + g) * 3 + kh) * 3 + kw] * sc[o * CIN + i];
    wpk[tid] = (__bf16)v;
}

// ---------------------------------------------------------------------------
// Kernel 3: WMMA implicit GEMM.
// 8 waves/block, each wave: one 16-wide M tile (16 consecutive w in one (b,h)
// row) x full N=64 (4 accumulators). A fragment reused across the 4 N tiles.
// A/B fragments loaded per the ISA 16-bit 16x32 lane layout: lane l<16 holds
// K {k0..k0+7, k0+16..k0+23}, lane l+16 holds {k0+8..k0+15, k0+24..k0+31} =>
// two aligned b128 loads per fragment from K-contiguous memory.
// ---------------------------------------------------------------------------
__global__ __launch_bounds__(256)
void kan_wmma_gemm(const __bf16* __restrict__ bases,
                   const __bf16* __restrict__ wpk,
                   float* __restrict__ out) {
    const int lane     = threadIdx.x & 31;
    const int wave     = threadIdx.x >> 5;
    const int laneLo   = lane & 15;
    const int laneHalf = lane >> 4;          // which K half this lane supplies

    const int mTile = blockIdx.x * 8 + wave; // 0..1023
    const int m0 = mTile << 4;
    const int b  = m0 >> 10;                 // batch
    const int hw = m0 & 1023;
    const int h  = hw >> 5;                  // output row
    const int w0 = hw & 31;                  // 0 or 16

    v8f acc[4] = {};                          // 16x64 f32 accumulators

    for (int kh = 0; kh < 3; ++kh) {
        for (int kw = 0; kw < 3; ++kw) {
            // A row for this lane: padded pixel (b, h+kh, w0+laneLo+kw)
            const __bf16* aRow =
                bases + (size_t)((b * HP + (h + kh)) * WP + (w0 + laneLo + kw)) * KPIX;
            const __bf16* bTap = wpk + (size_t)(kh * 3 + kw) * (COUT * KPIX);
#pragma unroll 4
            for (int k0 = 0; k0 < KPIX; k0 += 32) {
                union { v16bf v; v8bf h2[2]; } a;
                a.h2[0] = *(const v8bf*)(aRow + k0 + laneHalf * 8);
                a.h2[1] = *(const v8bf*)(aRow + k0 + 16 + laneHalf * 8);
#pragma unroll
                for (int nt = 0; nt < 4; ++nt) {
                    const __bf16* bCol = bTap + (size_t)(nt * 16 + laneLo) * KPIX;
                    union { v16bf v; v8bf h2[2]; } bb;
                    bb.h2[0] = *(const v8bf*)(bCol + k0 + laneHalf * 8);
                    bb.h2[1] = *(const v8bf*)(bCol + k0 + 16 + laneHalf * 8);
                    acc[nt] = __builtin_amdgcn_wmma_f32_16x16x32_bf16(
                        false, a.v, false, bb.v, (short)0, acc[nt], false, false);
                }
            }
        }
    }

    // Epilogue: C/D layout => lane holds N=laneLo, M = laneHalf*8 + r, and M
    // maps to consecutive w -> the 8 floats are contiguous in memory.
#pragma unroll
    for (int nt = 0; nt < 4; ++nt) {
        const int o = nt * 16 + laneLo;
        float* dst = out + (size_t)b * (COUT * HH * WW) + (size_t)o * (HH * WW)
                   + h * WW + w0 + laneHalf * 8;
        union { v8f v; v4f q[2]; } c;
        c.v = acc[nt];
        *(v4f*)(dst)     = c.q[0];
        *(v4f*)(dst + 4) = c.q[1];
    }
}

// ---------------------------------------------------------------------------
extern "C" void kernel_launch(void* const* d_in, const int* in_sizes, int n_in,
                              void* d_out, int out_size, void* d_ws, size_t ws_size,
                              hipStream_t stream) {
    const float* x  = (const float*)d_in[0];   // [16,64,32,32]
    const float* sw = (const float*)d_in[1];   // [64,64,8,3,3]
    const float* sc = (const float*)d_in[2];   // [64,64]
    float* out = (float*)d_out;                // [16,64,32,32]

    __bf16* bases = (__bf16*)d_ws;                                   // 18.94 MB
    __bf16* wpk   = bases + (size_t)BATCH * HP * WP * CIN * GS;      // +0.59 MB

    const int totB = BATCH * HP * WP * CIN;        // 1,183,744 threads
    kan_bases<<<(totB + 255) / 256, 256, 0, stream>>>(x, bases, totB);

    const int totW = 9 * COUT * KPIX;              // 294,912 threads
    kan_pack_w<<<(totW + 255) / 256, 256, 0, stream>>>(sw, sc, wpk, totW);

    // 1024 M-tiles / 8 waves per block
    kan_wmma_gemm<<<128, 256, 0, stream>>>(bases, wpk, out);
}
